// GeoConvNet3DPCSeg_28570122453850
// MI455X (gfx1250) — matrix operand order repaired and checked by reading, not compiled, not run on hardware
//
#include <hip/hip_runtime.h>

// ---------------------------------------------------------------------------
// PointNet++ segmentation pipeline for gfx1250 (MI455X).
// GEMMs: f32 in HBM, bf16 in LDS, v_wmma_f32_16x16x32_bf16, f32 accumulate.
// Conventions enforced by the host code:
//   - every GEMM row count E is a multiple of 128
//   - every GEMM input X has row stride ldX that is a multiple of 32, with
//     channels [Cin, ldX) zero-filled  ->  the k-loop is completely guard-free
//   - W reads use index-clamp + select-zero (no EXEC predication)
// ---------------------------------------------------------------------------

#define WMMA_BM 128
#define WMMA_BN 64
#define WMMA_BK 32
#define ASTR 40   // LDS row stride in bf16 elems (80B: 16B-aligned, padded)
#define BN_RSPLIT 64

typedef __attribute__((ext_vector_type(16))) __bf16 v16bf;
typedef __attribute__((ext_vector_type(8)))  __bf16 v8bf;
typedef __attribute__((ext_vector_type(8)))  float  v8f;

union V16U { v16bf v; v8bf h[2]; };

// f32 -> bf16 via C-level fptrunc (RNE); lets the backend pick a native
// v_cvt if gfx1250 has one, otherwise it emits the software RNE sequence.
__device__ __forceinline__ __bf16 f2bf(float f) {
  return (__bf16)f;
}

__device__ __forceinline__ unsigned pack_bf2(float a, float b) {
  unsigned short ua = __builtin_bit_cast(unsigned short, (__bf16)a);
  unsigned short ub = __builtin_bit_cast(unsigned short, (__bf16)b);
  return ((unsigned)ub << 16) | (unsigned)ua;
}

// ---------------------------------------------------------------------------
// Y[E x Cout] = X[E x Cin] @ W[Cin x Cout] (+ bias).
// ---------------------------------------------------------------------------
__global__ __launch_bounds__(256)
void gemm_bf16_wmma(const float* __restrict__ X, int ldX,
                    const float* __restrict__ W, const float* __restrict__ bias,
                    float* __restrict__ Y, int E, int Cin, int Cout) {
  __shared__ __attribute__((aligned(16))) __bf16 As[WMMA_BM][ASTR];
  __shared__ __attribute__((aligned(16))) __bf16 Bs[WMMA_BN][ASTR];

  const int tid  = threadIdx.x;
  const int lane = tid & 31;
  const int wid  = tid >> 5;
  const int wm   = wid & 3;    // wave position along M (4)
  const int wn   = wid >> 2;   // wave position along N (2)
  const int lh   = lane >> 4;  // lane half (0/1)
  const int l16  = lane & 15;
  const int rowBase = blockIdx.x * WMMA_BM;
  const int colBase = blockIdx.y * WMMA_BN;

  v8f zero = {0.f,0.f,0.f,0.f,0.f,0.f,0.f,0.f};
  v8f acc[2][2];
  acc[0][0] = zero; acc[0][1] = zero; acc[1][0] = zero; acc[1][1] = zero;

  for (int k0 = 0; k0 < ldX; k0 += WMMA_BK) {
    __syncthreads();
    // A tile: 4 float4 loads per thread, unconditional (row/k pad guaranteed).
#pragma unroll
    for (int it = 0; it < 4; ++it) {
      int e4 = (tid + it * 256) * 4;
      int r = e4 >> 5, c = e4 & 31;
      const float4 v = *(const float4*)(X + (long)(rowBase + r) * ldX + (k0 + c));
      unsigned lo = pack_bf2(v.x, v.y);
      unsigned hi = pack_bf2(v.z, v.w);
      *(unsigned long long*)&As[r][c] =
          ((unsigned long long)hi << 32) | (unsigned long long)lo;
    }
    // B tile transposed into Bs[n][k]; clamp+select instead of predication.
#pragma unroll
    for (int it = 0; it < 8; ++it) {
      int e = tid + it * 256;
      int nn = e & 63, kk = e >> 6;
      int gk = k0 + kk, gn = colBase + nn;
      int gkc = gk < Cin  ? gk : (Cin - 1);
      int gnc = gn < Cout ? gn : (Cout - 1);
      float v = W[(long)gkc * Cout + gnc];
      v = (gk < Cin && gn < Cout) ? v : 0.0f;
      Bs[nn][kk] = f2bf(v);
    }
    __syncthreads();

#pragma unroll
    for (int i = 0; i < 2; ++i) {
      int ar = wm * 32 + i * 16 + l16;
      V16U a;
      a.h[0] = *(const v8bf*)&As[ar][lh * 8];        // K = lh*8 .. +7
      a.h[1] = *(const v8bf*)&As[ar][16 + lh * 8];   // K = 16+lh*8 .. +7
#pragma unroll
      for (int j = 0; j < 2; ++j) {
        int bn = wn * 32 + j * 16 + l16;
        V16U bf;
        bf.h[0] = *(const v8bf*)&Bs[bn][lh * 8];       // K = lh*8 .. +7
        bf.h[1] = *(const v8bf*)&Bs[bn][16 + lh * 8];  // K = 16+lh*8 .. +7
        acc[i][j] = __builtin_amdgcn_wmma_f32_16x16x32_bf16(
            false, a.v, false, bf.v, (short)0, acc[i][j], false, false);
      }
    }
  }

#pragma unroll
  for (int i = 0; i < 2; ++i)
#pragma unroll
    for (int j = 0; j < 2; ++j)
#pragma unroll
      for (int v = 0; v < 8; ++v) {
        int r = rowBase + wm * 32 + i * 16 + lh * 8 + v;
        int c = colBase + wn * 32 + j * 16 + l16;
        if (c < Cout) {                       // rows always in range (E%128==0)
          float o = acc[i][j][v];
          if (bias) o += bias[c];
          Y[(long)r * Cout + c] = o;
        }
      }
}

// ---------------------------------------------------------------------------
// Masked BatchNorm statistics, two-phase for parallelism (deterministic).
// Phase 1: grid (ceil(C/64), BN_RSPLIT); 64 channels x 4 row-groups per block.
// ---------------------------------------------------------------------------
__global__ __launch_bounds__(256)
void bn_stats_part(const float* __restrict__ Y, const int* __restrict__ mask,
                   float* __restrict__ sPart, float* __restrict__ qPart,
                   float* __restrict__ cPart, int E, int C) {
  int tid = threadIdx.x;
  int c   = blockIdx.x * 64 + (tid & 63);
  int rg  = tid >> 6;               // 0..3
  int yb  = blockIdx.y;
  bool cok = c < C;
  float s = 0.f, q = 0.f, cnt = 0.f;
  for (int e = yb * 4 + rg; e < E; e += 4 * BN_RSPLIT) {
    float m = mask ? (mask[e] ? 1.0f : 0.0f) : 1.0f;
    float x = cok ? Y[(long)e * C + c] : 0.0f;
    s += x * m; q += x * x * m; cnt += m;
  }
  __shared__ float ss[256], qq[256], cc[4];
  ss[tid] = s; qq[tid] = q;
  if ((tid & 63) == 0) cc[rg] = cnt;
  __syncthreads();
  if (rg == 0) {
    s = ss[tid] + ss[tid + 64] + ss[tid + 128] + ss[tid + 192];
    q = qq[tid] + qq[tid + 64] + qq[tid + 128] + qq[tid + 192];
    if (cok) {
      sPart[(long)yb * C + c] = s;
      qPart[(long)yb * C + c] = q;
    }
    if (tid == 0 && blockIdx.x == 0) cPart[yb] = cc[0] + cc[1] + cc[2] + cc[3];
  }
}

__global__ __launch_bounds__(256)
void bn_finalize(const float* __restrict__ sPart, const float* __restrict__ qPart,
                 const float* __restrict__ cPart, float* __restrict__ mean,
                 float* __restrict__ rstd, int C) {
  int c = blockIdx.x * 256 + threadIdx.x;
  if (c >= C) return;
  float s = 0.f, q = 0.f, cnt = 0.f;
  for (int y = 0; y < BN_RSPLIT; ++y) {
    s += sPart[(long)y * C + c];
    q += qPart[(long)y * C + c];
    cnt += cPart[y];
  }
  float n  = fmaxf(cnt, 1.0f);
  float mu = s / n;
  float var = fmaxf(q / n - mu * mu, 0.0f);
  mean[c] = mu;
  rstd[c] = rsqrtf(var + 1e-5f);
}

__global__ void bn_apply_relu(float* __restrict__ Y, const float* __restrict__ g,
                              const float* __restrict__ b, const float* __restrict__ mean,
                              const float* __restrict__ rstd, long total, int C) {
  long i = (long)blockIdx.x * blockDim.x + threadIdx.x;
  if (i >= total) return;
  int c = (int)(i % C);
  float x = Y[i];
  x = g[c] * (x - mean[c]) * rstd[c] + b[c];
  Y[i] = fmaxf(x, 0.0f);
}

// ---------------------------------------------------------------------------
// Farthest point sampling: one block per batch (n % 256 == 0, n <= 2048).
// Coordinates + running distances live in registers; argmax per round is a
// 5-step __shfl_xor reduce + 8-entry LDS reduce (2 barriers per round).
// ---------------------------------------------------------------------------
__global__ __launch_bounds__(256)
void fps_kernel(const float* __restrict__ pos, int n, int m, int* __restrict__ idx_out) {
  int b = blockIdx.x;
  const float* P = pos + (long)b * n * 3;
  int* out = idx_out + (long)b * m;
  int tid = threadIdx.x, lane = tid & 31, wid = tid >> 5;
  int nE = n >> 8;
  float px[8], py[8], pz[8], dr[8];
  __shared__ float wv[8];
  __shared__ int   wi[8];
  __shared__ int   s_nxt;
  float x0 = P[0], y0 = P[1], z0 = P[2];
  float best = -1.0f; int bi = 0x7fffffff;
#pragma unroll
  for (int i = 0; i < 8; ++i) if (i < nE) {
    int e = tid + (i << 8);
    px[i] = P[e*3]; py[i] = P[e*3+1]; pz[i] = P[e*3+2];
    float dx = px[i]-x0, dy = py[i]-y0, dz = pz[i]-z0;
    float d = dx*dx + dy*dy + dz*dz;
    dr[i] = d;
    if (d > best) { best = d; bi = e; }
  }
  if (tid == 0) out[0] = 0;
  for (int it = 1; it < m; ++it) {
#pragma unroll
    for (int off = 16; off > 0; off >>= 1) {
      float ov = __shfl_xor(best, off, 32);
      int   oi = __shfl_xor(bi,   off, 32);
      if (ov > best || (ov == best && oi < bi)) { best = ov; bi = oi; }
    }
    if (lane == 0) { wv[wid] = best; wi[wid] = bi; }
    __syncthreads();
    if (tid == 0) {
      float fb = wv[0]; int fi = wi[0];
#pragma unroll
      for (int w = 1; w < 8; ++w)
        if (wv[w] > fb || (wv[w] == fb && wi[w] < fi)) { fb = wv[w]; fi = wi[w]; }
      out[it] = fi; s_nxt = fi;
    }
    __syncthreads();
    int nxt = s_nxt;
    float xn = P[nxt*3], yn = P[nxt*3+1], zn = P[nxt*3+2];
    best = -1.0f; bi = 0x7fffffff;
#pragma unroll
    for (int i = 0; i < 8; ++i) if (i < nE) {
      int e = tid + (i << 8);
      float dx = px[i]-xn, dy = py[i]-yn, dz = pz[i]-zn;
      float d = dx*dx + dy*dy + dz*dz;
      float nd = d < dr[i] ? d : dr[i];
      dr[i] = nd;
      if (nd > best) { best = nd; bi = e; }
    }
  }
}

// ---------------------------------------------------------------------------
// Radius grouping: one block per centroid; K argmin rounds, register-resident
// distances, shfl reduce (2 barriers per round).
// ---------------------------------------------------------------------------
__global__ __launch_bounds__(256)
void group_kernel(const float* __restrict__ pos, const int* __restrict__ fpsidx,
                  float* __restrict__ cent, int* __restrict__ nbr, int* __restrict__ valid,
                  int n, int M, int K, float r2) {
  int bm = blockIdx.x;     // [0, B*M)
  int b  = bm / M;
  int tid = threadIdx.x, lane = tid & 31, wid = tid >> 5;
  int nE = n >> 8;
  const float* P = pos + (long)b * n * 3;
  __shared__ float wv[8];
  __shared__ int   wi[8];
  __shared__ float s_mv;
  __shared__ int   s_mi;
  int ci = fpsidx[bm];
  float cx = P[ci*3], cy = P[ci*3+1], cz = P[ci*3+2];
  if (tid == 0) { cent[bm*3] = cx; cent[bm*3+1] = cy; cent[bm*3+2] = cz; }
  float dr[8];
#pragma unroll
  for (int i = 0; i < 8; ++i) dr[i] = 3.0e38f;
#pragma unroll
  for (int i = 0; i < 8; ++i) if (i < nE) {
    int e = tid + (i << 8);
    float dx = P[e*3]-cx, dy = P[e*3+1]-cy, dz = P[e*3+2]-cz;
    float d = dx*dx + dy*dy + dz*dz;
    dr[i] = (d <= r2) ? d : 3.0e38f;
  }
  for (int k = 0; k < K; ++k) {
    float best = 3.0e38f; int bi = 0x7fffffff;
#pragma unroll
    for (int i = 0; i < 8; ++i) if (i < nE) {
      int e = tid + (i << 8);
      if (dr[i] < best) { best = dr[i]; bi = e; }
    }
#pragma unroll
    for (int off = 16; off > 0; off >>= 1) {
      float ov = __shfl_xor(best, off, 32);
      int   oi = __shfl_xor(bi,   off, 32);
      if (ov < best || (ov == best && oi < bi)) { best = ov; bi = oi; }
    }
    if (lane == 0) { wv[wid] = best; wi[wid] = bi; }
    __syncthreads();
    if (tid == 0) {
      float fb = wv[0]; int fi = wi[0];
#pragma unroll
      for (int w = 1; w < 8; ++w)
        if (wv[w] < fb || (wv[w] == fb && wi[w] < fi)) { fb = wv[w]; fi = wi[w]; }
      int ok = fb < 3.0e38f;
      nbr[(long)bm * K + k]   = ok ? fi : 0;
      valid[(long)bm * K + k] = ok;
      s_mv = fb; s_mi = fi;
    }
    __syncthreads();
    int mi = s_mi; float mvv = s_mv;
    if (mvv < 3.0e38f && (mi & 255) == tid) {
      int slot = mi >> 8;
#pragma unroll
      for (int i = 0; i < 8; ++i) if (i == slot) dr[i] = 3.0e38f;
    }
  }
}

// ---------------------------------------------------------------------------
// Edge feature build: concat(x[nbr], pos[nbr]-centroid), pad to stride Cs.
// ---------------------------------------------------------------------------
__global__ void edge_feats(const float* __restrict__ pos, const float* __restrict__ x,
                           const float* __restrict__ cent, const int* __restrict__ nbr,
                           float* __restrict__ out, long Etot, int n, int M, int K,
                           int Cx, int Cs) {
  long e = (long)blockIdx.x * blockDim.x + threadIdx.x;
  if (e >= Etot) return;
  long bm = e / K;
  int  b  = (int)(bm / M);
  int  j  = nbr[e];
  int  Cf = Cx + 3;
  float* o = out + e * Cs;
  if (x) {
    const float* xs = x + ((long)b * n + j) * Cx;
    for (int c = 0; c < Cx; ++c) o[c] = xs[c];
  }
  const float* ps = pos + ((long)b * n + j) * 3;
  const float* cs = cent + bm * 3;
  o[Cx + 0] = ps[0] - cs[0];
  o[Cx + 1] = ps[1] - cs[1];
  o[Cx + 2] = ps[2] - cs[2];
  for (int c = Cf; c < Cs; ++c) o[c] = 0.0f;  // zero pad channels
}

// ---------------------------------------------------------------------------
// Max over K valid neighbors.
// ---------------------------------------------------------------------------
__global__ void maxpool_k(const float* __restrict__ h, const int* __restrict__ valid,
                          float* __restrict__ out, long BMtot, int K, int C) {
  long i = (long)blockIdx.x * blockDim.x + threadIdx.x;
  if (i >= BMtot * C) return;
  long bm = i / C; int c = (int)(i % C);
  float m = -3.0e38f;
  for (int k = 0; k < K; ++k) {
    if (valid[bm * K + k]) {
      float v = h[(bm * K + k) * C + c];
      if (v > m) m = v;
    }
  }
  out[bm * C + c] = m;
}

// ---------------------------------------------------------------------------
// kNN-3 inverse-square-distance interpolation + concat with fine features.
// ---------------------------------------------------------------------------
__global__ void fp_interp(const float* __restrict__ pos_f, const float* __restrict__ pos_c,
                          const float* __restrict__ x_c, int Cc,
                          const float* __restrict__ x_f, int Cf,
                          float* __restrict__ out, int nf, int nc) {
  long p = (long)blockIdx.x * blockDim.x + threadIdx.x;
  if (p >= (long)4 * nf) return;
  int b = (int)(p / nf);
  const float* pf = pos_f + p * 3;
  float px = pf[0], py = pf[1], pz = pf[2];
  float d0 = 3e38f, d1 = 3e38f, d2v = 3e38f;
  int i0 = 0, i1 = 0, i2 = 0;
  const float* pc = pos_c + (long)b * nc * 3;
  for (int j = 0; j < nc; ++j) {
    float dx = px - pc[j*3], dy = py - pc[j*3+1], dz = pz - pc[j*3+2];
    float d = dx*dx + dy*dy + dz*dz;
    if (d < d0)      { d2v = d1; i2 = i1; d1 = d0; i1 = i0; d0 = d; i0 = j; }
    else if (d < d1) { d2v = d1; i2 = i1; d1 = d;  i1 = j; }
    else if (d < d2v){ d2v = d;  i2 = j; }
  }
  float w0 = 1.0f / fmaxf(d0, 1e-16f);
  float w1 = 1.0f / fmaxf(d1, 1e-16f);
  float w2 = 1.0f / fmaxf(d2v, 1e-16f);
  float ws = w0 + w1 + w2; w0 /= ws; w1 /= ws; w2 /= ws;
  int Ct = Cc + Cf;
  float* o = out + p * Ct;
  const float* xc = x_c + (long)b * nc * Cc;
  for (int c = 0; c < Cc; ++c)
    o[c] = w0 * xc[i0*Cc + c] + w1 * xc[i1*Cc + c] + w2 * xc[i2*Cc + c];
  if (Cf > 0) {
    const float* xf = x_f + p * Cf;
    for (int c = 0; c < Cf; ++c) o[Cc + c] = xf[c];
  }
}

// ---------------------------------------------------------------------------
// Host orchestration
// ---------------------------------------------------------------------------
static void mlp_layer(hipStream_t s, const float* X, int ldX, const float* W,
                      const float* g, const float* b, const int* mask, float* Y,
                      float* mean, float* rstd, float* sPart, float* qPart, float* cPart,
                      long E, int Cin, int Cout) {
  dim3 grid((unsigned)(E / WMMA_BM), (unsigned)((Cout + WMMA_BN - 1) / WMMA_BN));
  gemm_bf16_wmma<<<grid, 256, 0, s>>>(X, ldX, W, (const float*)nullptr, Y, (int)E, Cin, Cout);
  dim3 sgrid((Cout + 63) / 64, BN_RSPLIT);
  bn_stats_part<<<sgrid, 256, 0, s>>>(Y, mask, sPart, qPart, cPart, (int)E, Cout);
  bn_finalize<<<(Cout + 255) / 256, 256, 0, s>>>(sPart, qPart, cPart, mean, rstd, Cout);
  long total = E * Cout;
  bn_apply_relu<<<(unsigned)((total + 255) / 256), 256, 0, s>>>(Y, g, b, mean, rstd, total, Cout);
}

extern "C" void kernel_launch(void* const* d_in, const int* in_sizes, int n_in,
                              void* d_out, int out_size, void* d_ws, size_t ws_size,
                              hipStream_t stream) {
  const int Bc = 4, Np = 2048, Kg = 64, NC = 50;
  const int M1 = 1024, M2 = 256, M3 = 64;
  const long E1 = (long)Bc * M1 * Kg;  // 262144
  const long E2 = (long)Bc * M2 * Kg;  // 65536
  const long E3 = (long)Bc * M3 * Kg;  // 16384

  // ---- input pointer mapping (detect flatten scheme via pos' unique size) ----
  auto F = [&](int i) { return (const float*)d_in[i]; };
  int posIdx = -1;
  for (int i = 0; i < n_in; ++i) if (in_sizes[i] == Bc * Np * 3) { posIdx = i; break; }

  const float *pos = nullptr;
  const float *saW[3][3], *saG[3][3], *saB[3][3];
  const float *fpW[3][2], *fpG[3][2], *fpB[3][2];   // index 0=fp1, 1=fp2, 2=fp3
  const float *hW1, *hW2, *hG, *hB, *hBias;

  if (posIdx == 0) {
    // insertion order: pos, sa1, sa2, sa3, fp3, fp2, fp1, head; per layer W,g,b
    pos = F(0);
    int i = 1;
    for (int ss = 0; ss < 3; ++ss)
      for (int l = 0; l < 3; ++l) { saW[ss][l] = F(i++); saG[ss][l] = F(i++); saB[ss][l] = F(i++); }
    for (int f = 2; f >= 0; --f)  // fp3, fp2, fp1
      for (int l = 0; l < 2; ++l) { fpW[f][l] = F(i++); fpG[f][l] = F(i++); fpB[f][l] = F(i++); }
    hW1 = F(i++); hG = F(i++); hB = F(i++); hW2 = F(i++); hBias = F(i++);
  } else {
    // sorted keys: fp1,fp2,fp3,head,sa1,sa2,sa3, then pos; per layer W,b,g; head W1,W2,b,bias,g
    int i = 0;
    for (int f = 0; f < 3; ++f)
      for (int l = 0; l < 2; ++l) { fpW[f][l] = F(i++); fpB[f][l] = F(i++); fpG[f][l] = F(i++); }
    hW1 = F(i++); hW2 = F(i++); hB = F(i++); hBias = F(i++); hG = F(i++);
    for (int ss = 0; ss < 3; ++ss)
      for (int l = 0; l < 3; ++l) { saW[ss][l] = F(i++); saB[ss][l] = F(i++); saG[ss][l] = F(i++); }
    pos = F(i);
  }

  // ---- workspace carve ----
  char* wsp = (char*)d_ws;
  size_t off = 0;
  auto alloc = [&](size_t nbytes) -> void* {
    void* p = (void*)(wsp + off);
    off += (nbytes + 255) & ~(size_t)255;
    return p;
  };
  int*   idx1 = (int*)  alloc((size_t)Bc * M1 * 4);
  float* pos1 = (float*)alloc((size_t)Bc * M1 * 3 * 4);
  int*   nbr1 = (int*)  alloc((size_t)E1 * 4);
  int*   val1 = (int*)  alloc((size_t)E1 * 4);
  int*   idx2 = (int*)  alloc((size_t)Bc * M2 * 4);
  float* pos2 = (float*)alloc((size_t)Bc * M2 * 3 * 4);
  int*   nbr2 = (int*)  alloc((size_t)E2 * 4);
  int*   val2 = (int*)  alloc((size_t)E2 * 4);
  int*   idx3 = (int*)  alloc((size_t)Bc * M3 * 4);
  float* pos3 = (float*)alloc((size_t)Bc * M3 * 3 * 4);
  int*   nbr3 = (int*)  alloc((size_t)E3 * 4);
  int*   val3 = (int*)  alloc((size_t)E3 * 4);
  float* x1   = (float*)alloc((size_t)Bc * M1 * 128 * 4);
  float* x2   = (float*)alloc((size_t)Bc * M2 * 256 * 4);
  float* x3   = (float*)alloc((size_t)Bc * M3 * 512 * 4);
  float* x2u  = (float*)alloc((size_t)Bc * M2 * 256 * 4);
  float* x1u  = (float*)alloc((size_t)Bc * M1 * 128 * 4);
  float* x0u  = (float*)alloc((size_t)Bc * Np * 128 * 4);
  float* mean = (float*)alloc(512 * 4);
  float* rstd = (float*)alloc(512 * 4);
  float* sPart = (float*)alloc((size_t)BN_RSPLIT * 512 * 4);
  float* qPart = (float*)alloc((size_t)BN_RSPLIT * 512 * 4);
  float* cPart = (float*)alloc((size_t)BN_RSPLIT * 4);
  float* bufA = (float*)alloc((size_t)E1 * 64 * 4);   // 64 MB
  float* bufB = (float*)alloc((size_t)E1 * 128 * 4);  // 128 MB
  (void)ws_size; (void)out_size; (void)n_in;

  hipStream_t s = stream;

  // ================= SA1: pos (2048) -> 1024 pts, feats 3->64->64->128 =======
  fps_kernel<<<Bc, 256, 0, s>>>(pos, Np, M1, idx1);
  group_kernel<<<Bc * M1, 256, 0, s>>>(pos, idx1, pos1, nbr1, val1, Np, M1, Kg, 0.04f);
  edge_feats<<<(unsigned)((E1 + 255) / 256), 256, 0, s>>>(pos, nullptr, pos1, nbr1, bufA, E1, Np, M1, Kg, 0, 32);
  mlp_layer(s, bufA,  32, saW[0][0], saG[0][0], saB[0][0], val1, bufB, mean, rstd, sPart, qPart, cPart, E1, 3, 64);
  mlp_layer(s, bufB,  64, saW[0][1], saG[0][1], saB[0][1], val1, bufA, mean, rstd, sPart, qPart, cPart, E1, 64, 64);
  mlp_layer(s, bufA,  64, saW[0][2], saG[0][2], saB[0][2], val1, bufB, mean, rstd, sPart, qPart, cPart, E1, 64, 128);
  maxpool_k<<<(unsigned)(((long)Bc * M1 * 128 + 255) / 256), 256, 0, s>>>(bufB, val1, x1, (long)Bc * M1, Kg, 128);

  // ================= SA2: 1024 -> 256 pts, feats 131->128->128->256 ==========
  fps_kernel<<<Bc, 256, 0, s>>>(pos1, M1, M2, idx2);
  group_kernel<<<Bc * M2, 256, 0, s>>>(pos1, idx2, pos2, nbr2, val2, M1, M2, Kg, 0.16f);
  edge_feats<<<(unsigned)((E2 + 255) / 256), 256, 0, s>>>(pos1, x1, pos2, nbr2, bufA, E2, M1, M2, Kg, 128, 160);
  mlp_layer(s, bufA, 160, saW[1][0], saG[1][0], saB[1][0], val2, bufB, mean, rstd, sPart, qPart, cPart, E2, 131, 128);
  mlp_layer(s, bufB, 128, saW[1][1], saG[1][1], saB[1][1], val2, bufA, mean, rstd, sPart, qPart, cPart, E2, 128, 128);
  mlp_layer(s, bufA, 128, saW[1][2], saG[1][2], saB[1][2], val2, bufB, mean, rstd, sPart, qPart, cPart, E2, 128, 256);
  maxpool_k<<<(unsigned)(((long)Bc * M2 * 256 + 255) / 256), 256, 0, s>>>(bufB, val2, x2, (long)Bc * M2, Kg, 256);

  // ================= SA3: 256 -> 64 pts, feats 259->256->256->512 ============
  fps_kernel<<<Bc, 256, 0, s>>>(pos2, M2, M3, idx3);
  group_kernel<<<Bc * M3, 256, 0, s>>>(pos2, idx3, pos3, nbr3, val3, M2, M3, Kg, 0.64f);
  edge_feats<<<(unsigned)((E3 + 255) / 256), 256, 0, s>>>(pos2, x2, pos3, nbr3, bufA, E3, M2, M3, Kg, 256, 288);
  mlp_layer(s, bufA, 288, saW[2][0], saG[2][0], saB[2][0], val3, bufB, mean, rstd, sPart, qPart, cPart, E3, 259, 256);
  mlp_layer(s, bufB, 256, saW[2][1], saG[2][1], saB[2][1], val3, bufA, mean, rstd, sPart, qPart, cPart, E3, 256, 256);
  mlp_layer(s, bufA, 256, saW[2][2], saG[2][2], saB[2][2], val3, bufB, mean, rstd, sPart, qPart, cPart, E3, 256, 512);
  maxpool_k<<<(unsigned)(((long)Bc * M3 * 512 + 255) / 256), 256, 0, s>>>(bufB, val3, x3, (long)Bc * M3, Kg, 512);

  // ================= FP3: interp x3 @ pos2, concat x2 -> 768->256->256 =======
  long P2 = (long)Bc * M2;
  fp_interp<<<(unsigned)((P2 + 255) / 256), 256, 0, s>>>(pos2, pos3, x3, 512, x2, 256, bufA, M2, M3);
  mlp_layer(s, bufA, 768, fpW[2][0], fpG[2][0], fpB[2][0], nullptr, bufB, mean, rstd, sPart, qPart, cPart, P2, 768, 256);
  mlp_layer(s, bufB, 256, fpW[2][1], fpG[2][1], fpB[2][1], nullptr, x2u, mean, rstd, sPart, qPart, cPart, P2, 256, 256);

  // ================= FP2: interp x2u @ pos1, concat x1 -> 384->128->128 ======
  long P1 = (long)Bc * M1;
  fp_interp<<<(unsigned)((P1 + 255) / 256), 256, 0, s>>>(pos1, pos2, x2u, 256, x1, 128, bufA, M1, M2);
  mlp_layer(s, bufA, 384, fpW[1][0], fpG[1][0], fpB[1][0], nullptr, bufB, mean, rstd, sPart, qPart, cPart, P1, 384, 128);
  mlp_layer(s, bufB, 128, fpW[1][1], fpG[1][1], fpB[1][1], nullptr, x1u, mean, rstd, sPart, qPart, cPart, P1, 128, 128);

  // ================= FP1: interp x1u @ pos -> 128->128->128 ==================
  long P0 = (long)Bc * Np;
  fp_interp<<<(unsigned)((P0 + 255) / 256), 256, 0, s>>>(pos, pos1, x1u, 128, nullptr, 0, bufA, Np, M1);
  mlp_layer(s, bufA, 128, fpW[0][0], fpG[0][0], fpB[0][0], nullptr, bufB, mean, rstd, sPart, qPart, cPart, P0, 128, 128);
  mlp_layer(s, bufB, 128, fpW[0][1], fpG[0][1], fpB[0][1], nullptr, x0u, mean, rstd, sPart, qPart, cPart, P0, 128, 128);

  // ================= Head: 128->128 (BN,ReLU) -> 128->50 (+bias) =============
  {
    dim3 g1((unsigned)(P0 / WMMA_BM), (128 + WMMA_BN - 1) / WMMA_BN);
    gemm_bf16_wmma<<<g1, 256, 0, s>>>(x0u, 128, hW1, (const float*)nullptr, bufA, (int)P0, 128, 128);
    dim3 sg((128 + 63) / 64, BN_RSPLIT);
    bn_stats_part<<<sg, 256, 0, s>>>(bufA, nullptr, sPart, qPart, cPart, (int)P0, 128);
    bn_finalize<<<1, 256, 0, s>>>(sPart, qPart, cPart, mean, rstd, 128);
    long tot = P0 * 128;
    bn_apply_relu<<<(unsigned)((tot + 255) / 256), 256, 0, s>>>(bufA, hG, hB, mean, rstd, tot, 128);
    dim3 g2((unsigned)(P0 / WMMA_BM), (NC + WMMA_BN - 1) / WMMA_BN);
    gemm_bf16_wmma<<<g2, 256, 0, s>>>(bufA, 128, hW2, hBias, (float*)d_out, (int)P0, 128, NC);
  }
}
